// DualStreamNetwork_68247030333826
// MI455X (gfx1250) — compile-verified, matrix operand order checked by reading, not run on hardware
//
#include <hip/hip_runtime.h>
#include <hip/hip_bf16.h>

// ---------------------------------------------------------------------------
// Types for WMMA
// ---------------------------------------------------------------------------
typedef __attribute__((ext_vector_type(16))) _Float16 v16h;
typedef __attribute__((ext_vector_type(8)))  _Float16 v8h;
typedef __attribute__((ext_vector_type(8)))  float    v8f;

#define B_SZ     64
#define T_SZ     64
#define E_SZ     512
#define D_MODEL  1024
#define D_INNER  2048
#define D_STATE  16
#define DT_RANK  64
#define M_ROWS   (B_SZ * T_SZ)   // 4096

// ---------------------------------------------------------------------------
// WMMA fragment load: 16x32 f16 tile, row-major source with leading dim `ld`.
// ISA layout (16-bit A / transposed-B): lane L -> row (L&15),
//   k-chunks at (L>>4)*8 and (L>>4)*8 + 16, 8 halfs (16B) each.
// ---------------------------------------------------------------------------
__device__ __forceinline__ v16h load_frag(const _Float16* __restrict__ base,
                                          int row, int kb, int ld, int k) {
  const _Float16* p = base + (size_t)row * ld + k + kb;
  v8h lo = *(const v8h*)(p);
  v8h hi = *(const v8h*)(p + 16);
  v16h r;
#pragma unroll
  for (int i = 0; i < 8; ++i) { r[i] = lo[i]; r[i + 8] = hi[i]; }
  return r;
}

// Epilogue variants
#define EPI_F32            0
#define EPI_F16            1
#define EPI_BIAS_RELU_F16  2
#define EPI_BIAS_SOFTP_F16 3

template <int EPI>
__device__ __forceinline__ void store_tile(void* __restrict__ Dv,
                                           const float* __restrict__ bias,
                                           v8f acc, int m0, int n0, int N,
                                           int lane) {
  const int n     = n0 + (lane & 15);
  const int mbase = m0 + (lane >> 4) * 8;   // VGPR r: M = r (+8 for hi lanes)
  float bv = 0.f;
  if (EPI == EPI_BIAS_RELU_F16 || EPI == EPI_BIAS_SOFTP_F16) bv = bias[n];
#pragma unroll
  for (int r = 0; r < 8; ++r) {
    float v = acc[r];
    size_t off = (size_t)(mbase + r) * (size_t)N + (size_t)n;
    if (EPI == EPI_F32) {
      ((float*)Dv)[off] = v;
    } else if (EPI == EPI_F16) {
      ((_Float16*)Dv)[off] = (_Float16)v;
    } else if (EPI == EPI_BIAS_RELU_F16) {
      v += bv;
      v = v > 0.f ? v : 0.f;
      ((_Float16*)Dv)[off] = (_Float16)v;
    } else { // bias + softplus
      v += bv;
      float sp = (v > 20.f) ? v : log1pf(__expf(v));
      ((_Float16*)Dv)[off] = (_Float16)sp;
    }
  }
}

// ---------------------------------------------------------------------------
// Tiled WMMA GEMM: D(MxN) = A(MxK,f16,row-major) * Bt(NxK,f16,row-major)^T
// 256 threads = 8 wave32. Each wave owns a (MT*16) x (NT*16) tile of D
// (MT x NT WMMA fragments -> MT*NT WMMAs per K-step of 32, MT+NT frag loads).
// Block tile: (WM*MT*16) x (WN*NT*16), WM*WN == 8.
// K is compile-time: exact trip count, no guard branch, unroll-2 pipelining.
// ---------------------------------------------------------------------------
template <int WM, int WN, int MT, int NT, int K, int EPI>
__global__ __launch_bounds__(256) void gemm_wmma(
    const _Float16* __restrict__ A, const _Float16* __restrict__ Bt,
    const float* __restrict__ bias, void* __restrict__ Dv, int N) {
  const int lane = threadIdx.x & 31;
  const int wave = threadIdx.x >> 5;
  const int wm = wave % WM;
  const int wn = wave / WM;
  const int m0 = blockIdx.y * (WM * MT * 16) + wm * (MT * 16);
  const int n0 = blockIdx.x * (WN * NT * 16) + wn * (NT * 16);

  const int row = lane & 15;
  const int kb  = (lane >> 4) * 8;

  const _Float16* Abase = A  + (size_t)m0 * K;
  const _Float16* Bbase = Bt + (size_t)n0 * K;

  v8f zero = {};
  v8f acc[MT][NT];
#pragma unroll
  for (int i = 0; i < MT; ++i)
#pragma unroll
    for (int j = 0; j < NT; ++j) acc[i][j] = zero;

#pragma unroll 2
  for (int k = 0; k < K; k += 32) {
    v16h af[MT], bf[NT];
#pragma unroll
    for (int i = 0; i < MT; ++i)
      af[i] = load_frag(Abase + (size_t)(i * 16) * K, row, kb, K, k);
#pragma unroll
    for (int j = 0; j < NT; ++j)
      bf[j] = load_frag(Bbase + (size_t)(j * 16) * K, row, kb, K, k);
#pragma unroll
    for (int i = 0; i < MT; ++i)
#pragma unroll
      for (int j = 0; j < NT; ++j)
        acc[i][j] = __builtin_amdgcn_wmma_f32_16x16x32_f16(
            false, af[i], false, bf[j], (short)0, acc[i][j], false, false);
  }

#pragma unroll
  for (int i = 0; i < MT; ++i)
#pragma unroll
    for (int j = 0; j < NT; ++j)
      store_tile<EPI>(Dv, bias, acc[i][j], m0 + i * 16, n0 + j * 16, N, lane);
}

// ---------------------------------------------------------------------------
// Elementwise / data-prep kernels
// ---------------------------------------------------------------------------

// concat(spatial, temporal) -> f16 A matrix (4096 x 1024)
__global__ void concat_cvt_kernel(const float* __restrict__ sp,
                                  const float* __restrict__ tp,
                                  _Float16* __restrict__ dst) {
  size_t i = (size_t)blockIdx.x * blockDim.x + threadIdx.x; // 4096*1024
  int col   = (int)(i & (D_MODEL - 1));
  size_t rt = i >> 10;
  float v = (col < E_SZ) ? sp[rt * E_SZ + col] : tp[rt * E_SZ + (col - E_SZ)];
  dst[i] = (_Float16)v;
}

// f32 KxN weight -> f16 NxK (transposed) for B-fragment contiguity
__global__ void transpose_cvt_kernel(const float* __restrict__ src,
                                     _Float16* __restrict__ dst,
                                     int K, int N) {
  size_t i = (size_t)blockIdx.x * blockDim.x + threadIdx.x; // N*K
  int n = (int)(i / (size_t)K);
  int k = (int)(i % (size_t)K);
  dst[i] = (_Float16)src[(size_t)k * N + n];
}

// strided f32 slice -> packed f16 (dt_raw = proj[:, :64])
__global__ void slice_cvt_kernel(const float* __restrict__ src,
                                 _Float16* __restrict__ dst,
                                 int src_ld, int cols) {
  size_t i = (size_t)blockIdx.x * blockDim.x + threadIdx.x;
  int c = (int)(i % (size_t)cols);
  size_t r = i / (size_t)cols;
  dst[i] = (_Float16)src[r * src_ld + c];
}

// causal depthwise conv1d (K=4) + bias + SiLU; reads xi = xz[:, :2048]
__global__ void conv_silu_kernel(const _Float16* __restrict__ xz,
                                 const float* __restrict__ w,
                                 const float* __restrict__ bias,
                                 _Float16* __restrict__ xc) {
  size_t i = (size_t)blockIdx.x * blockDim.x + threadIdx.x; // 4096*2048
  int d     = (int)(i & (D_INNER - 1));
  size_t rt = i >> 11;            // b*T + t
  int t = (int)(rt & (T_SZ - 1));
  float acc = bias[d];
#pragma unroll
  for (int k = 0; k < 4; ++k) {
    int tt = t - 3 + k;
    if (tt >= 0)
      acc += w[d * 4 + k] * (float)xz[(rt - 3 + k) * (2 * D_INNER) + d];
  }
  float s = acc / (1.f + __expf(-acc));
  xc[i] = (_Float16)s;
}

// Selective scan: one thread per (b, d); 16-state recurrence in registers.
// Per-timestep B/C (32 floats, shared by whole block) is staged into LDS with
// CDNA5 async global->LDS copies (tracked by ASYNCcnt), then broadcast-read.
// Fuses D_skip residual and SiLU(z) gating; writes f16 y (A of W_out GEMM).
__global__ __launch_bounds__(256) void scan_kernel(
    const _Float16* __restrict__ dt, const _Float16* __restrict__ xc,
    const _Float16* __restrict__ xz, const float* __restrict__ proj,
    const float* __restrict__ A_log, const float* __restrict__ D_skip,
    _Float16* __restrict__ y) {
  __shared__ float sBC[2 * D_STATE];   // [0:16)=B_t, [16:32)=C_t
  const int b = blockIdx.x >> 3;                       // 8 blocks per batch
  const int d = ((blockIdx.x & 7) << 8) + threadIdx.x; // 0..2047

  float A[D_STATE], h[D_STATE];
#pragma unroll
  for (int n = 0; n < D_STATE; ++n) {
    A[n] = -__expf(A_log[d * D_STATE + n]);
    h[n] = 0.f;
  }
  const float Dv = D_skip[d];

  for (int t = 0; t < T_SZ; ++t) {
    size_t rt = (size_t)b * T_SZ + t;
    if (threadIdx.x < 8) {
      // 8 lanes x 16B async copies: proj[rt, 64:96] -> LDS, no VGPR bounce
      unsigned lds_off = (unsigned)(uintptr_t)(&sBC[threadIdx.x * 4]);
      const float* gp =
          proj + rt * (DT_RANK + 2 * D_STATE) + DT_RANK + threadIdx.x * 4;
      asm volatile("global_load_async_to_lds_b128 %0, %1, off"
                   :: "v"(lds_off), "v"(gp) : "memory");
      asm volatile("s_wait_asynccnt 0x0" ::: "memory");
    }
    __syncthreads();
    float dtv = (float)dt[rt * D_INNER + d];
    float xv  = (float)xc[rt * D_INNER + d];
    float dtx = dtv * xv;
    float yv = 0.f;
#pragma unroll
    for (int n = 0; n < D_STATE; ++n) {
      h[n] = __expf(dtv * A[n]) * h[n] + dtx * sBC[n];
      yv += h[n] * sBC[D_STATE + n];
    }
    float zv  = (float)xz[rt * (2 * D_INNER) + D_INNER + d];
    float g   = zv / (1.f + __expf(-zv));
    float out = (yv + xv * Dv) * g;
    y[rt * D_INNER + d] = (_Float16)out;
    __syncthreads();
  }
}

// final tiny head: out(4096x6) = h2(4096x256,f16) @ W3(256x6,f32) + b3
__global__ void head_kernel(const _Float16* __restrict__ h2,
                            const float* __restrict__ W3,
                            const float* __restrict__ b3,
                            float* __restrict__ out) {
  int i = blockIdx.x * blockDim.x + threadIdx.x; // 24576
  int c = i % 6;
  int r = i / 6;
  float acc = b3[c];
  for (int k = 0; k < 256; ++k)
    acc += (float)h2[r * 256 + k] * W3[k * 6 + c];
  out[i] = acc;
}

// ---------------------------------------------------------------------------
// Host launch
// ---------------------------------------------------------------------------
extern "C" void kernel_launch(void* const* d_in, const int* in_sizes, int n_in,
                              void* d_out, int out_size, void* d_ws,
                              size_t ws_size, hipStream_t stream) {
  (void)in_sizes; (void)n_in; (void)out_size; (void)ws_size;

  const float* spatial  = (const float*)d_in[0];
  const float* temporal = (const float*)d_in[1];
  const float* W_in   = (const float*)d_in[2];
  const float* conv_w = (const float*)d_in[3];
  const float* conv_b = (const float*)d_in[4];
  const float* W_x    = (const float*)d_in[5];
  const float* W_dt   = (const float*)d_in[6];
  const float* b_dt   = (const float*)d_in[7];
  const float* A_log  = (const float*)d_in[8];
  const float* D_skip = (const float*)d_in[9];
  const float* W_out  = (const float*)d_in[10];
  const float* W1     = (const float*)d_in[11];
  const float* b1     = (const float*)d_in[12];
  const float* W2     = (const float*)d_in[13];
  const float* b2     = (const float*)d_in[14];
  const float* W3     = (const float*)d_in[15];
  const float* b3     = (const float*)d_in[16];
  float* out = (float*)d_out;

  // workspace carve-up (f16 buffers except f32 proj; 256B aligned)
  char* ws = (char*)d_ws;
  size_t off = 0;
  auto alloc = [&](size_t bytes) {
    void* p = ws + off;
    off += (bytes + 255) & ~(size_t)255;
    return p;
  };
  _Float16* xA     = (_Float16*)alloc((size_t)M_ROWS * D_MODEL * 2);      // 8 MB
  _Float16* Win_t  = (_Float16*)alloc((size_t)(2 * D_INNER) * D_MODEL * 2); // 8 MB
  _Float16* xz     = (_Float16*)alloc((size_t)M_ROWS * 2 * D_INNER * 2);  // 32 MB
  _Float16* xc     = (_Float16*)alloc((size_t)M_ROWS * D_INNER * 2);      // 16 MB
  _Float16* Wx_t   = (_Float16*)alloc((size_t)96 * D_INNER * 2);
  float*    proj   = (float*)   alloc((size_t)M_ROWS * 96 * 4);
  _Float16* dtraw  = (_Float16*)alloc((size_t)M_ROWS * DT_RANK * 2);
  _Float16* Wdt_t  = (_Float16*)alloc((size_t)D_INNER * DT_RANK * 2);
  _Float16* dtbuf  = (_Float16*)alloc((size_t)M_ROWS * D_INNER * 2);      // 16 MB
  _Float16* ybuf   = (_Float16*)alloc((size_t)M_ROWS * D_INNER * 2);      // 16 MB
  _Float16* Wout_t = (_Float16*)alloc((size_t)D_MODEL * D_INNER * 2);     // 4 MB
  _Float16* fbuf   = (_Float16*)alloc((size_t)M_ROWS * D_MODEL * 2);      // 8 MB
  _Float16* W1_t   = (_Float16*)alloc((size_t)512 * D_MODEL * 2);
  _Float16* h1buf  = (_Float16*)alloc((size_t)M_ROWS * 512 * 2);
  _Float16* W2_t   = (_Float16*)alloc((size_t)256 * 512 * 2);
  _Float16* h2buf  = (_Float16*)alloc((size_t)M_ROWS * 256 * 2);

  // 1) input concat -> f16
  concat_cvt_kernel<<<(M_ROWS * D_MODEL) / 256, 256, 0, stream>>>(
      spatial, temporal, xA);

  // 2) weight transposes (f32 KxN -> f16 NxK)
  transpose_cvt_kernel<<<(D_MODEL * 2 * D_INNER) / 256, 256, 0, stream>>>(
      W_in, Win_t, D_MODEL, 2 * D_INNER);
  transpose_cvt_kernel<<<(D_INNER * 96) / 256, 256, 0, stream>>>(
      W_x, Wx_t, D_INNER, 96);
  transpose_cvt_kernel<<<(DT_RANK * D_INNER) / 256, 256, 0, stream>>>(
      W_dt, Wdt_t, DT_RANK, D_INNER);
  transpose_cvt_kernel<<<(D_INNER * D_MODEL) / 256, 256, 0, stream>>>(
      W_out, Wout_t, D_INNER, D_MODEL);
  transpose_cvt_kernel<<<(D_MODEL * 512) / 256, 256, 0, stream>>>(
      W1, W1_t, D_MODEL, 512);
  transpose_cvt_kernel<<<(512 * 256) / 256, 256, 0, stream>>>(
      W2, W2_t, 512, 256);

  // 3) xz = x @ W_in          (4096 x 4096 x 1024), 128x128 block tiles
  gemm_wmma<4, 2, 2, 4, D_MODEL, EPI_F16>
      <<<dim3(4096 / 128, 4096 / 128), 256, 0, stream>>>(
          xA, Win_t, nullptr, xz, 2 * D_INNER);

  // 4) causal conv + SiLU -> xc
  conv_silu_kernel<<<(M_ROWS * D_INNER) / 256, 256, 0, stream>>>(
      xz, conv_w, conv_b, xc);

  // 5) proj = xc @ W_x        (4096 x 96 x 2048), 256x48 block tiles
  gemm_wmma<8, 1, 2, 3, D_INNER, EPI_F32>
      <<<dim3(96 / 48, 4096 / 256), 256, 0, stream>>>(
          xc, Wx_t, nullptr, proj, 96);

  // 6) dt_raw slice -> f16
  slice_cvt_kernel<<<(M_ROWS * DT_RANK) / 256, 256, 0, stream>>>(
      proj, dtraw, 96, DT_RANK);

  // 7) dt = softplus(dt_raw @ W_dt + b_dt)   (4096 x 2048 x 64)
  gemm_wmma<4, 2, 2, 4, DT_RANK, EPI_BIAS_SOFTP_F16>
      <<<dim3(2048 / 128, 4096 / 128), 256, 0, stream>>>(
          dtraw, Wdt_t, b_dt, dtbuf, D_INNER);

  // 8) selective scan + D-skip + SiLU(z) gating -> y
  scan_kernel<<<B_SZ * 8, 256, 0, stream>>>(dtbuf, xc, xz, proj, A_log,
                                            D_skip, ybuf);

  // 9) f = y @ W_out          (4096 x 1024 x 2048)
  gemm_wmma<4, 2, 2, 4, D_INNER, EPI_F16>
      <<<dim3(1024 / 128, 4096 / 128), 256, 0, stream>>>(
          ybuf, Wout_t, nullptr, fbuf, D_MODEL);

  // 10) h1 = relu(f @ W1 + b1)   (4096 x 512 x 1024)
  gemm_wmma<4, 2, 2, 4, D_MODEL, EPI_BIAS_RELU_F16>
      <<<dim3(512 / 128, 4096 / 128), 256, 0, stream>>>(
          fbuf, W1_t, b1, h1buf, 512);

  // 11) h2 = relu(h1 @ W2 + b2)  (4096 x 256 x 512)
  gemm_wmma<4, 2, 2, 4, 512, EPI_BIAS_RELU_F16>
      <<<dim3(256 / 128, 4096 / 128), 256, 0, stream>>>(
          h1buf, W2_t, b2, h2buf, 256);

  // 12) out = h2 @ W3 + b3       (4096 x 6 x 256)
  head_kernel<<<(M_ROWS * 6) / 256, 256, 0, stream>>>(h2buf, W3, b3, out);
}
